// GaussianProjectionIntegration_9285719294014
// MI455X (gfx1250) — matrix operand order repaired and checked by reading, not compiled, not run on hardware
//
#include <hip/hip_runtime.h>
#include <math.h>

typedef __attribute__((ext_vector_type(16))) _Float16 v16h;
typedef __attribute__((ext_vector_type(8)))  float    v8f;

static constexpr int   HH = 512, WW = 512;
static constexpr int   NN = 8192, FF = 2, NB = 8;
static constexpr int   MM = NN * FF;           // 16384 separable gaussians
static constexpr int   KT = MM / 32;           // 512 k-tiles of 32
static constexpr int   HT = HH / 16;           // 32 row tiles
static constexpr int   WT = WW / 16;           // 32 col tiles
static constexpr int   KSPLIT = 8;
static constexpr float A2P = 1.0f;             // 1/PIXEL_SIZE
static constexpr float INV_AREA = 1.0f;
static constexpr float SQRT2 = 1.4142135623730951f;

// ---------------- zero output ----------------
__global__ void zero_kernel(float* __restrict__ out, int n) {
  int i = blockIdx.x * blockDim.x + threadIdx.x;
  if (i < n) out[i] = 0.0f;
}

// ---------------- 4x4 inverse (Gauss-Jordan, 1 thread per batch) ----------------
__global__ void invert4x4_kernel(const float* __restrict__ Tm, float* __restrict__ Minv) {
  int b = threadIdx.x;
  if (b >= NB) return;
  float a[4][4], inv[4][4];
  for (int i = 0; i < 4; ++i)
    for (int j = 0; j < 4; ++j) {
      a[i][j] = Tm[b * 16 + i * 4 + j];
      inv[i][j] = (i == j) ? 1.0f : 0.0f;
    }
  for (int col = 0; col < 4; ++col) {
    int piv = col; float best = fabsf(a[col][col]);
    for (int r = col + 1; r < 4; ++r) { float v = fabsf(a[r][col]); if (v > best) { best = v; piv = r; } }
    if (piv != col)
      for (int j = 0; j < 4; ++j) {
        float t = a[col][j]; a[col][j] = a[piv][j]; a[piv][j] = t;
        t = inv[col][j]; inv[col][j] = inv[piv][j]; inv[piv][j] = t;
      }
    float d = 1.0f / a[col][col];
    for (int j = 0; j < 4; ++j) { a[col][j] *= d; inv[col][j] *= d; }
    for (int r = 0; r < 4; ++r) if (r != col) {
      float f = a[r][col];
      for (int j = 0; j < 4; ++j) { a[r][j] -= f * a[col][j]; inv[r][j] -= f * inv[col][j]; }
    }
  }
  for (int i = 0; i < 16; ++i) Minv[b * 16 + i] = inv[i / 4][i % 4];
}

// ---------------- projection: cp[b][n] = (Minv[b]·[c,1]).xy / .w * A2P ----------------
__global__ void project_kernel(const float* __restrict__ Minv, const float* __restrict__ centers,
                               float* __restrict__ cp) {
  int idx = blockIdx.x * blockDim.x + threadIdx.x;
  if (idx >= NB * NN) return;
  int b = idx / NN, n = idx % NN;
  const float* Mi = Minv + b * 16;
  float x = centers[n * 3 + 0], y = centers[n * 3 + 1], z = centers[n * 3 + 2];
  float p0 = Mi[0]  * x + Mi[1]  * y + Mi[2]  * z + Mi[3];
  float p1 = Mi[4]  * x + Mi[5]  * y + Mi[6]  * z + Mi[7];
  float p3 = Mi[12] * x + Mi[13] * y + Mi[14] * z + Mi[15];
  float invw = A2P / p3;
  cp[idx * 2 + 0] = p0 * invw;
  cp[idx * 2 + 1] = p1 * invw;
}

// ---------------- A profile: weights*ph*INV_AREA in WMMA A-fragment layout ----------------
// flat half index = (kt*HT + ht)*512 + lane*16 + half
// A 16x32 f16 layout: lanes 0-15 hold K 0-7 (halves 0-7) & 16-23 (halves 8-15);
//                     lanes 16-31 hold K 8-15 & 24-31; row M = lane&15.
__global__ void profileA_kernel(const float* __restrict__ cp, const float* __restrict__ scales,
                                const float* __restrict__ weights, _Float16* __restrict__ aBuf, int b) {
  int idx = blockIdx.x * blockDim.x + threadIdx.x;   // M*H threads
  int tile = idx >> 9, lane = (idx >> 4) & 31, half = idx & 15;
  int kt = tile >> 5, ht = tile & 31;
  int km = (half & 7) | ((lane >> 4) << 3) | ((half >> 3) << 4);
  int m  = (kt << 5) | km;
  int h  = (ht << 4) | (lane & 15);
  int n  = m >> 1;
  float s   = scales[m] * A2P;
  float inv = 1.0f / (s * SQRT2);
  float c   = cp[((size_t)b * NN + n) * 2 + 0];
  float x   = (float)h;
  float ph  = 0.5f * (erff((x + 0.5f - c) * inv) - erff((x - 0.5f - c) * inv));
  aBuf[idx] = (_Float16)(weights[m] * ph * INV_AREA);
}

// ---------------- B profile: pw in WMMA B-fragment layout ----------------
// B 32x16 f16 layout: lane = col + 16*((K>>3)&1), half = (K&7) + 8*(K>=16)
__global__ void profileB_kernel(const float* __restrict__ cp, const float* __restrict__ scales,
                                _Float16* __restrict__ bBuf, int b) {
  int idx = blockIdx.x * blockDim.x + threadIdx.x;   // M*W threads
  int tile = idx >> 9, lane = (idx >> 4) & 31, half = idx & 15;
  int kt = tile >> 5, wt = tile & 31;
  int km = (half & 7) | ((lane >> 4) << 3) | ((half >> 3) << 4);
  int m  = (kt << 5) | km;
  int w  = (wt << 4) | (lane & 15);
  int n  = m >> 1;
  float s   = scales[m] * A2P;
  float inv = 1.0f / (s * SQRT2);
  float c   = cp[((size_t)b * NN + n) * 2 + 1];
  float x   = (float)w;
  bBuf[idx] = (_Float16)(0.5f * (erff((x + 0.5f - c) * inv) - erff((x - 0.5f - c) * inv)));
}

// ---------------- WMMA GEMM: out[b] += A(H x M) * B(M x W), K split 8 ways ----------------
__global__ void __launch_bounds__(32)
gemm_wmma_kernel(const _Float16* __restrict__ aBuf, const _Float16* __restrict__ bBuf,
                 float* __restrict__ out, int b) {
  const int lane = threadIdx.x;
  const int wt0  = blockIdx.x * 2;          // 2 col tiles  (32 cols)
  const int ht0  = blockIdx.y * 2;          // 2 row tiles  (32 rows)
  const int kc   = blockIdx.z;              // k chunk
  v8f acc00 = {}, acc01 = {}, acc10 = {}, acc11 = {};
  const int ktBeg = kc * (KT / KSPLIT);
  const int ktEnd = ktBeg + (KT / KSPLIT);
  const int lo = lane << 4;                 // 16 halves (32B) per lane per fragment
  for (int kt = ktBeg; kt < ktEnd; ++kt) {
    const v16h a0 = *(const v16h*)(aBuf + (((size_t)(kt * HT + ht0))     << 9) + lo);
    const v16h a1 = *(const v16h*)(aBuf + (((size_t)(kt * HT + ht0 + 1)) << 9) + lo);
    const v16h b0 = *(const v16h*)(bBuf + (((size_t)(kt * WT + wt0))     << 9) + lo);
    const v16h b1 = *(const v16h*)(bBuf + (((size_t)(kt * WT + wt0 + 1)) << 9) + lo);
    acc00 = __builtin_amdgcn_wmma_f32_16x16x32_f16(false, a0, false, b0, (short)0, acc00, false, false);
    acc01 = __builtin_amdgcn_wmma_f32_16x16x32_f16(false, a0, false, b1, (short)0, acc01, false, false);
    acc10 = __builtin_amdgcn_wmma_f32_16x16x32_f16(false, a1, false, b0, (short)0, acc10, false, false);
    acc11 = __builtin_amdgcn_wmma_f32_16x16x32_f16(false, a1, false, b1, (short)0, acc11, false, false);
  }
  // C/D f32 layout: VGPR i -> row i + 8*(lane>=16), col = lane&15
  const int rlo = (lane >> 4) << 3;
  const int col = lane & 15;
  float* outb = out + (size_t)b * HH * WW;
  for (int i = 0; i < 8; ++i) {
    int h0 = (ht0 << 4) + rlo + i;
    int h1 = ((ht0 + 1) << 4) + rlo + i;
    atomicAdd(&outb[(size_t)h0 * WW + (wt0 << 4) + col],        acc00[i]);
    atomicAdd(&outb[(size_t)h0 * WW + ((wt0 + 1) << 4) + col],  acc01[i]);
    atomicAdd(&outb[(size_t)h1 * WW + (wt0 << 4) + col],        acc10[i]);
    atomicAdd(&outb[(size_t)h1 * WW + ((wt0 + 1) << 4) + col],  acc11[i]);
  }
}

// ---------------- bilinear point-mass splat ----------------
__global__ void splat_kernel(const float* __restrict__ cp, const float* __restrict__ bw,
                             float* __restrict__ out) {
  int idx = blockIdx.x * blockDim.x + threadIdx.x;
  if (idx >= NB * NN) return;
  int b = idx / NN, n = idx % NN;
  float r = cp[idx * 2 + 0], c = cp[idx * 2 + 1];
  float r0f = floorf(r), c0f = floorf(c);
  float fr = r - r0f, fc = c - c0f;
  int r0 = (int)r0f, c0 = (int)c0f;
  float w = bw[n] * INV_AREA;
  float* outb = out + (size_t)b * HH * WW;
  float wts[4] = { (1.f - fr) * (1.f - fc), (1.f - fr) * fc, fr * (1.f - fc), fr * fc };
  int drs[4] = {0, 0, 1, 1}, dcs[4] = {0, 1, 0, 1};
  for (int k = 0; k < 4; ++k) {
    int rr = min(max(r0 + drs[k], 0), HH - 1);
    int cc = min(max(c0 + dcs[k], 0), WW - 1);
    atomicAdd(&outb[(size_t)rr * WW + cc], w * wts[k]);
  }
}

extern "C" void kernel_launch(void* const* d_in, const int* in_sizes, int n_in,
                              void* d_out, int out_size, void* d_ws, size_t ws_size,
                              hipStream_t stream) {
  const float* Tm      = (const float*)d_in[0];  // (B,4,4)
  const float* centers = (const float*)d_in[1];  // (N,3)
  const float* scales  = (const float*)d_in[2];  // (N,F)
  const float* weights = (const float*)d_in[3];  // (N,F)
  const float* bweights= (const float*)d_in[4];  // (N,)
  float* out = (float*)d_out;                    // (B,H,W) f32
  (void)in_sizes; (void)n_in; (void)out_size; (void)ws_size;

  char* ws = (char*)d_ws;
  float*    Minv = (float*)ws;                         // 8*16 f32
  float*    cp   = (float*)(ws + 1024);                // B*N*2 f32  (512 KiB)
  _Float16* aBuf = (_Float16*)(ws + (1u << 20));       // M*H f16    (16 MiB), per-batch
  _Float16* bBuf = (_Float16*)(ws + (1u << 20) + (size_t)MM * HH * 2); // M*W f16 (16 MiB)

  zero_kernel<<<(NB * HH * WW + 255) / 256, 256, 0, stream>>>(out, NB * HH * WW);
  invert4x4_kernel<<<1, 32, 0, stream>>>(Tm, Minv);
  project_kernel<<<(NB * NN + 255) / 256, 256, 0, stream>>>(Minv, centers, cp);

  const int profBlocks = (MM * HH) / 256;              // exact (32768)
  for (int b = 0; b < NB; ++b) {
    profileA_kernel<<<profBlocks, 256, 0, stream>>>(cp, scales, weights, aBuf, b);
    profileB_kernel<<<profBlocks, 256, 0, stream>>>(cp, scales, bBuf, b);
    dim3 grid(WW / 32, HH / 32, KSPLIT);               // (16,16,8), 1 wave per block
    gemm_wmma_kernel<<<grid, 32, 0, stream>>>(aBuf, bBuf, out, b);
  }
  splat_kernel<<<(NB * NN + 255) / 256, 256, 0, stream>>>(cp, bweights, out);
}
